// ArHmmLm_19353122636097
// MI455X (gfx1250) — matrix operand order repaired and checked by reading, not compiled, not run on hardware
//
#include <hip/hip_runtime.h>
#include <hip/hip_bf16.h>

typedef __attribute__((ext_vector_type(16))) __bf16 v16bf;
typedef __attribute__((ext_vector_type(4)))  __bf16 v4bf;
typedef __attribute__((ext_vector_type(8)))  float  v8f;
typedef __attribute__((ext_vector_type(4)))  float  v4f;

#define BB   4
#define TT   64
#define VV   8192
#define CC   64
#define HH   256
#define NR   252            // B*(T-1) valid rows
#define NRP  256            // padded rows
#define VCHUNKS 128
#define VPC  (VV / VCHUNKS) // 64 v per chunk

// ---------------- conv front-end: h0 = relu(e_prev@W0 + e_cur@W1 + b) --------
__global__ __launch_bounds__(256) void conv_kernel(
    const int* __restrict__ text, const float* __restrict__ embW,
    const float* __restrict__ W0, const float* __restrict__ W1,
    const float* __restrict__ bias, float* __restrict__ h0)
{
  int r = blockIdx.x;             // 0..255 (252 valid)
  int col = threadIdx.x;
  if (r >= NR) { h0[r * HH + col] = 0.f; return; }
  int b = r / 63, t = r % 63;
  int tp = (t == 0) ? 0 : text[b * TT + t - 1];
  int tc = text[b * TT + t];
  __shared__ float ep[HH], ec[HH];
  ep[col] = embW[(long)tp * HH + col];
  ec[col] = embW[(long)tc * HH + col];
  __syncthreads();
  float acc = bias[col];
  for (int k = 0; k < HH; ++k)
    acc = fmaf(ep[k], W0[k * HH + col], fmaf(ec[k], W1[k * HH + col], acc));
  h0[r * HH + col] = fmaxf(acc, 0.f);
}

// ---------------- generic per-row residual MLP: relu(relu(x@W1+b1)@W2+b2)+x --
__global__ __launch_bounds__(256) void residual_kernel(
    const float* __restrict__ x, const float* __restrict__ W1,
    const float* __restrict__ b1, const float* __restrict__ W2,
    const float* __restrict__ b2, float* __restrict__ out)
{
  int r = blockIdx.x, col = threadIdx.x;
  __shared__ float xs[HH], t1[HH];
  xs[col] = x[r * HH + col];
  __syncthreads();
  float a = b1[col];
  for (int k = 0; k < HH; ++k) a = fmaf(xs[k], W1[k * HH + col], a);
  t1[col] = fmaxf(a, 0.f);
  __syncthreads();
  float a2 = b2[col];
  for (int k = 0; k < HH; ++k) a2 = fmaf(t1[k], W2[k * HH + col], a2);
  out[r * HH + col] = fmaxf(a2, 0.f) + xs[col];
}

// ---------------- start head: spre[r] = sres[r]·sW + sb ----------------------
__global__ __launch_bounds__(256) void start_head(
    const float* __restrict__ sres, const float* __restrict__ sW,
    const float* __restrict__ sb, float* __restrict__ spre)
{
  int r = blockIdx.x, tid = threadIdx.x;
  __shared__ float red[HH];
  red[tid] = sres[r * HH + tid] * sW[tid];
  __syncthreads();
  for (int off = 128; off > 0; off >>= 1) {
    if (tid < off) red[tid] += red[tid + off];
    __syncthreads();
  }
  if (tid == 0) spre[r] = red[0] + sb[0];
}

__global__ __launch_bounds__(64) void lsm64_kernel(
    const float* __restrict__ in, float* __restrict__ out)
{
  __shared__ float sh[CC];
  int i = threadIdx.x;
  sh[i] = in[i];
  __syncthreads();
  float m = -1e30f;
  for (int j = 0; j < CC; ++j) m = fmaxf(m, sh[j]);
  float s = 0.f;
  for (int j = 0; j < CC; ++j) s += __expf(sh[j] - m);
  out[i] = sh[i] - (m + __logf(s));
}

// ---------------- transition head: transT[i][j] = lsm_j(tres[j]@tW+tb)[i] ----
__global__ __launch_bounds__(64) void trans_head(
    const float* __restrict__ tres, const float* __restrict__ tW,
    const float* __restrict__ tb, float* __restrict__ transT)
{
  int j = blockIdx.x, i = threadIdx.x;  // row j (prev state), col i (next)
  __shared__ float xs[HH];
  __shared__ float rowv[CC];
  for (int k = i; k < HH; k += CC) xs[k] = tres[j * HH + k];
  __syncthreads();
  float acc = tb[i];
  for (int k = 0; k < HH; ++k) acc = fmaf(xs[k], tW[k * CC + i], acc);
  rowv[i] = acc;
  __syncthreads();
  float m = -1e30f;
  for (int k = 0; k < CC; ++k) m = fmaxf(m, rowv[k]);
  float s = 0.f;
  for (int k = 0; k < CC; ++k) s += __expf(rowv[k] - m);
  transT[i * CC + j] = acc - (m + __logf(s));
}

// ---------------- big WMMA kernel: partial sum_v exp(h·projW[v*64+c]) --------
// grid = VCHUNKS blocks of 512 threads (16 waves); wave w owns M-rows
// [16w,16w+16). One v-tile of proj_W (64 rows x 256 K) is staged in LDS as
// bf16, double-buffered: NT global loads for tile v+1 issue before the WMMA
// loop on tile v, stores to the other buffer follow, one barrier/iteration.
// proj_W (512 MB) is streamed from HBM exactly once, bypassing L2 (TH=NT).
// LDS layout uses an XOR swizzle k ^= (row&15)<<4 (chunk-preserving) to kill
// bank conflicts without padding: 2*64*256*2B = 64 KB exactly.
__global__ __launch_bounds__(512) void emis_partial_kernel(
    const float* __restrict__ projW, const float* __restrict__ hbuf,
    float* __restrict__ partial)
{
  __shared__ __attribute__((aligned(32))) __bf16 Bl[2][CC * HH];
  const int tid  = threadIdx.x;
  const int lane = tid & 31;
  const int wave = tid >> 5;        // 0..15 = M row tile
  const int mrow = lane & 15;
  const int half = lane >> 4;

  // A matrix (16x256 bf16): VGPR i holds K pair
  // half*16 + 2i within each K=32 slice (matches B's contiguous order).
  const float* hrow = hbuf + (wave * 16 + mrow) * HH;
  v16bf A[8];
#pragma unroll
  for (int s = 0; s < 8; ++s) {
    v16bf a;
#pragma unroll
    for (int i = 0; i < 8; ++i) {
      int kb = s * 32 + half * 16 + 2 * i;
      a[2 * i]     = (__bf16)hrow[kb];
      a[2 * i + 1] = (__bf16)hrow[kb + 1];
    }
    A[s] = a;
  }

  v8f sum[4];
#pragma unroll
  for (int ct = 0; ct < 4; ++ct)
#pragma unroll
    for (int e = 0; e < 8; ++e) sum[ct][e] = 0.f;

  const long vbase = (long)blockIdx.x * VPC;
  const v4f* srcBase = (const v4f*)projW + vbase * (CC * HH / 4);

  // preload tile 0 into buffer 0
#pragma unroll
  for (int j = 0; j < 8; ++j) {
    int i4 = j * 512 + tid;                 // float4 index in tile (0..4095)
    v4f f = __builtin_nontemporal_load(srcBase + i4);
    int e0 = i4 * 4;
    int row = e0 >> 8, k = e0 & 255;
    int ks = k ^ ((row & 15) << 4);         // chunk-preserving XOR swizzle
    v4bf p; p[0] = (__bf16)f[0]; p[1] = (__bf16)f[1];
    p[2] = (__bf16)f[2]; p[3] = (__bf16)f[3];
    *reinterpret_cast<v4bf*>(&Bl[0][row * HH + ks]) = p;
  }
  __syncthreads();

  for (int v = 0; v < VPC; ++v) {
    const int cur = v & 1;
    // stage next tile into registers (issues NT loads ahead of the math)
    v4f nx[8];
    if (v + 1 < VPC) {
      const v4f* src = srcBase + (long)(v + 1) * (CC * HH / 4);
#pragma unroll
      for (int j = 0; j < 8; ++j)
        nx[j] = __builtin_nontemporal_load(src + j * 512 + tid);
    }

    // compute on current buffer
#pragma unroll
    for (int ct = 0; ct < 4; ++ct) {
      v8f acc;
#pragma unroll
      for (int e = 0; e < 8; ++e) acc[e] = 0.f;
      const int row = ct * 16 + mrow;       // B row (= state c)
      const int swz = (row & 15) << 4;
      const __bf16* bp = &Bl[cur][row * HH];
#pragma unroll
      for (int s = 0; s < 8; ++s) {
        // B k order chosen here (same as A): VGPR i, lane l: N=l%16, K=(l/16)*16 + 2i,2i+1 per slice
        int cb = (half * 16 + s * 32) ^ swz;
        v16bf bmat = *reinterpret_cast<const v16bf*>(bp + cb);
        acc = __builtin_amdgcn_wmma_f32_16x16x32_bf16(
            false, A[s], false, bmat, (short)0, acc, false, false);
      }
#pragma unroll
      for (int e = 0; e < 8; ++e) sum[ct][e] += __expf(acc[e]);
    }

    // convert + store next tile into the other buffer
    if (v + 1 < VPC) {
#pragma unroll
      for (int j = 0; j < 8; ++j) {
        int i4 = j * 512 + tid;
        int e0 = i4 * 4;
        int row = e0 >> 8, k = e0 & 255;
        int ks = k ^ ((row & 15) << 4);
        v4bf p; p[0] = (__bf16)nx[j][0]; p[1] = (__bf16)nx[j][1];
        p[2] = (__bf16)nx[j][2]; p[3] = (__bf16)nx[j][3];
        *reinterpret_cast<v4bf*>(&Bl[cur ^ 1][row * HH + ks]) = p;
      }
    }
    __syncthreads();
  }

  // C/D layout: VGPR e -> M = e + 8*half, N = lane%16
#pragma unroll
  for (int ct = 0; ct < 4; ++ct)
#pragma unroll
    for (int e = 0; e < 8; ++e) {
      int r_out = wave * 16 + e + 8 * half;
      int c = ct * 16 + (lane & 15);
      partial[(long)blockIdx.x * (NRP * CC) + r_out * CC + c] = sum[ct][e];
    }
}

// ---------------- picked logit at the observed token -------------------------
__global__ __launch_bounds__(64) void picked_kernel(
    const float* __restrict__ hbuf, const float* __restrict__ projW,
    const int* __restrict__ text, float* __restrict__ picked)
{
  int r = blockIdx.x;                     // 0..251
  int b = r / 63, t = r % 63;
  int obs = text[b * TT + t + 1];
  __shared__ float hs[HH];
  for (int k = threadIdx.x; k < HH; k += 64) hs[k] = hbuf[r * HH + k];
  __syncthreads();
  int c = threadIdx.x;
  const float* wrow = projW + ((long)obs * CC + c) * HH;
  float acc = 0.f;
  for (int k = 0; k < HH; ++k) acc = fmaf(hs[k], wrow[k], acc);
  picked[r * CC + c] = acc;
}

// ---------------- em = picked - log(sum of chunk partials) -------------------
__global__ __launch_bounds__(256) void em_combine(
    const float* __restrict__ picked, const float* __restrict__ partial,
    float* __restrict__ em)
{
  int idx = blockIdx.x * blockDim.x + threadIdx.x;
  if (idx >= NR * CC) return;
  int r = idx >> 6, c = idx & 63;
  float s = 0.f;
  for (int ch = 0; ch < VCHUNKS; ++ch)
    s += partial[(long)ch * (NRP * CC) + r * CC + c];
  em[idx] = picked[idx] - __logf(s);
}

// ---------------- HMM forward-backward, evidence + elbo per batch ------------
__global__ __launch_bounds__(64) void hmm_kernel(
    const float* __restrict__ em, const float* __restrict__ transT,
    const float* __restrict__ startv, const int* __restrict__ lengths,
    float* __restrict__ ev_out, float* __restrict__ elbo_out)
{
  __shared__ float tr[CC * CC];
  __shared__ float alpha[63][CC];
  __shared__ float gcur[CC], gnext[CC], Zi[CC], red[CC], st[CC];
  const int b = blockIdx.x, i = threadIdx.x;
  for (int k = i; k < CC * CC; k += CC) tr[k] = transT[k];
  st[i] = startv[i];
  __syncthreads();
  int L = lengths[b];
  int idxT = L - 2; idxT = max(0, min(62, idxT));
  const float* emb_ = em + b * 63 * CC;
  { // t = 0 : alpha0[i] = em[0,i] + lse_j(tr[i][j] + start[j])
    float m = -1e30f;
    for (int j = 0; j < CC; ++j) m = fmaxf(m, tr[i * CC + j] + st[j]);
    float s = 0.f;
    for (int j = 0; j < CC; ++j) s += __expf(tr[i * CC + j] + st[j] - m);
    alpha[0][i] = emb_[i] + m + __logf(s);
  }
  __syncthreads();
  for (int t = 1; t <= idxT; ++t) {
    float m = -1e30f;
    for (int j = 0; j < CC; ++j) m = fmaxf(m, tr[i * CC + j] + alpha[t - 1][j]);
    float s = 0.f;
    for (int j = 0; j < CC; ++j) s += __expf(tr[i * CC + j] + alpha[t - 1][j] - m);
    alpha[t][i] = emb_[t * CC + i] + m + __logf(s);
    __syncthreads();
  }
  red[i] = alpha[idxT][i];
  __syncthreads();
  float m = -1e30f;
  for (int j = 0; j < CC; ++j) m = fmaxf(m, red[j]);
  float s = 0.f;
  for (int j = 0; j < CC; ++j) s += __expf(red[j] - m);
  float logZ = m + __logf(s);
  if (i == 0) ev_out[b] = logZ;
  gcur[i] = __expf(alpha[idxT][i] - logZ);
  __syncthreads();
  float elbo = 0.f;
  for (int t = idxT; t >= 1; --t) {
    float em_ti = emb_[t * CC + i];
    float Z = alpha[t][i] - em_ti;        // = lse_j(tr[i][j]+alpha[t-1][j])
    Zi[i] = Z;
    __syncthreads();
    float gi = gcur[i];
    for (int j = 0; j < CC; ++j) {
      float p = __expf(tr[i * CC + j] + alpha[t - 1][j] - Z);
      elbo += gi * p * (tr[i * CC + j] + em_ti);
    }
    float gn = 0.f;                        // this thread acts as state j=i
    for (int k2 = 0; k2 < CC; ++k2)
      gn += gcur[k2] * __expf(tr[k2 * CC + i] + alpha[t - 1][i] - Zi[k2]);
    gnext[i] = gn;
    __syncthreads();
    gcur[i] = gnext[i];
    __syncthreads();
  }
  { // t = 0
    float em0 = emb_[i];
    float Z0 = alpha[0][i] - em0;
    float gi = gcur[i];
    for (int j = 0; j < CC; ++j) {
      float p = __expf(tr[i * CC + j] + st[j] - Z0);
      elbo += gi * p * (tr[i * CC + j] + em0 + st[j]);
    }
  }
  red[i] = elbo;
  __syncthreads();
  for (int off = 32; off > 0; off >>= 1) {
    if (i < off) red[i] += red[i + off];
    __syncthreads();
  }
  if (i == 0) elbo_out[b] = red[0];
}

__global__ void finalize_kernel(const float* __restrict__ ev,
                                const float* __restrict__ el,
                                float* __restrict__ out)
{
  if (threadIdx.x == 0 && blockIdx.x == 0) {
    float a = 0.f, e = 0.f;
    for (int i = 0; i < BB; ++i) { a += el[i]; e += ev[i]; }
    out[0] = a;   // elbo
    out[1] = e;   // evidence
  }
}

// ---------------------------------------------------------------------------
extern "C" void kernel_launch(void* const* d_in, const int* in_sizes, int n_in,
                              void* d_out, int out_size, void* d_ws, size_t ws_size,
                              hipStream_t stream) {
  (void)in_sizes; (void)n_in; (void)out_size; (void)ws_size;
  const int*   text      = (const int*)  d_in[0];
  const int*   lengths   = (const int*)  d_in[1];
  const float* embW      = (const float*)d_in[2];
  const float* convW0    = (const float*)d_in[3];
  const float* convW1    = (const float*)d_in[4];
  const float* convB     = (const float*)d_in[5];
  const float* mW1       = (const float*)d_in[6];
  const float* mb1       = (const float*)d_in[7];
  const float* mW2       = (const float*)d_in[8];
  const float* mb2       = (const float*)d_in[9];
  const float* projW     = (const float*)d_in[10];
  const float* start_emb = (const float*)d_in[11];
  const float* sW1       = (const float*)d_in[12];
  const float* sb1       = (const float*)d_in[13];
  const float* sW2       = (const float*)d_in[14];
  const float* sb2       = (const float*)d_in[15];
  const float* s_out_W   = (const float*)d_in[16];
  const float* s_out_b   = (const float*)d_in[17];
  const float* state_emb = (const float*)d_in[18];
  const float* tW1       = (const float*)d_in[19];
  const float* tb1       = (const float*)d_in[20];
  const float* tW2       = (const float*)d_in[21];
  const float* tb2       = (const float*)d_in[22];
  const float* t_out_W   = (const float*)d_in[23];
  const float* t_out_b   = (const float*)d_in[24];
  float* out = (float*)d_out;

  float* ws = (float*)d_ws;
  float* ws_h       = ws;                        // 256*256
  float* ws_sres    = ws_h    + NRP * HH;        // 64*256
  float* ws_tres    = ws_sres + CC * HH;         // 64*256
  float* ws_spre    = ws_tres + CC * HH;         // 64
  float* ws_start   = ws_spre + CC;              // 64
  float* ws_trans   = ws_start + CC;             // 64*64
  float* ws_picked  = ws_trans + CC * CC;        // 256*64
  float* ws_em      = ws_picked + NRP * CC;      // 256*64
  float* ws_ev      = ws_em + NRP * CC;          // 4
  float* ws_elbo    = ws_ev + BB;                // 4
  float* ws_partial = ws_elbo + BB;              // VCHUNKS*256*64 (8 MB)

  conv_kernel<<<NRP, 256, 0, stream>>>(text, embW, convW0, convW1, convB, ws_h);
  residual_kernel<<<NRP, 256, 0, stream>>>(ws_h, mW1, mb1, mW2, mb2, ws_h);
  residual_kernel<<<CC, 256, 0, stream>>>(start_emb, sW1, sb1, sW2, sb2, ws_sres);
  residual_kernel<<<CC, 256, 0, stream>>>(state_emb, tW1, tb1, tW2, tb2, ws_tres);
  start_head<<<CC, 256, 0, stream>>>(ws_sres, s_out_W, s_out_b, ws_spre);
  lsm64_kernel<<<1, 64, 0, stream>>>(ws_spre, ws_start);
  trans_head<<<CC, 64, 0, stream>>>(ws_tres, t_out_W, t_out_b, ws_trans);

  emis_partial_kernel<<<VCHUNKS, 512, 0, stream>>>(projW, ws_h, ws_partial);
  picked_kernel<<<NR, 64, 0, stream>>>(ws_h, projW, text, ws_picked);
  em_combine<<<(NR * CC + 255) / 256, 256, 0, stream>>>(ws_picked, ws_partial, ws_em);

  hmm_kernel<<<BB, 64, 0, stream>>>(ws_em, ws_trans, ws_start, lengths, ws_ev, ws_elbo);
  finalize_kernel<<<1, 1, 0, stream>>>(ws_ev, ws_elbo, out);
}
